// RecurrentNeuralNetworkScratch_44908178047578
// MI455X (gfx1250) — compile-verified, matrix operand order checked by reading, not run on hardware
//
#include <hip/hip_runtime.h>

// RNN scratch for MI455X (gfx1250, wave32):
//   K0: transpose Wh -> WhT (fp32, one-time, 1MB)
//   K1: Xpre = x @ Wx^T           via v_wmma_f32_16x16x32_bf16 (fp32 accum)
//   K2: sequential scan (1 WGP, 1024 threads, fp32 critical path, bf16 H out)
//   K3: y = H @ Wy^T + by         via v_wmma_f32_16x16x32_bf16

#define SEQ_LEN 32768
#define IN_SZ   256
#define HID     512
#define OUT_SZ  256

typedef __attribute__((ext_vector_type(16))) __bf16 v16bf;
typedef __attribute__((ext_vector_type(8)))  __bf16 v8bf;
typedef __attribute__((ext_vector_type(8)))  float  v8f;

__device__ __forceinline__ __bf16 f2bf(float f) { return (__bf16)f; }

// Build one 16-element bf16 fragment (A-row or B-column) from an fp32 matrix.
// Per the CDNA5 16-bit 16x32 layout: lane L holds row/col (L%16); element e
// maps to K = ko + 16*(e/8) + 8*(L/16) + (e%8)  -> two contiguous 8-float runs.
__device__ __forceinline__ v16bf frag_f32(const float* __restrict__ base, int ld,
                                          int rc, int ko, int hi) {
  const float* p = base + (size_t)rc * ld + ko + hi * 8;
  float4 a0 = *(const float4*)(p);
  float4 a1 = *(const float4*)(p + 4);
  float4 b0 = *(const float4*)(p + 16);
  float4 b1 = *(const float4*)(p + 20);
  v16bf f;
  f[0]  = f2bf(a0.x); f[1]  = f2bf(a0.y); f[2]  = f2bf(a0.z); f[3]  = f2bf(a0.w);
  f[4]  = f2bf(a1.x); f[5]  = f2bf(a1.y); f[6]  = f2bf(a1.z); f[7]  = f2bf(a1.w);
  f[8]  = f2bf(b0.x); f[9]  = f2bf(b0.y); f[10] = f2bf(b0.z); f[11] = f2bf(b0.w);
  f[12] = f2bf(b1.x); f[13] = f2bf(b1.y); f[14] = f2bf(b1.z); f[15] = f2bf(b1.w);
  return f;
}

// Same fragment from a bf16 matrix: two 16-byte loads per lane.
__device__ __forceinline__ v16bf frag_bf16(const __bf16* __restrict__ base, int ld,
                                           int rc, int ko, int hi) {
  const __bf16* p = base + (size_t)rc * ld + ko + hi * 8;
  v8bf lo = *(const v8bf*)(p);
  v8bf hh = *(const v8bf*)(p + 16);
  v16bf f;
#pragma unroll
  for (int e = 0; e < 8; ++e) { f[e] = lo[e]; f[e + 8] = hh[e]; }
  return f;
}

// ---------------- K0: Wh -> WhT (fp32) ----------------
__global__ void wh_transpose(const float* __restrict__ Wh, float* __restrict__ WhT) {
  int i = blockIdx.x * blockDim.x + threadIdx.x;   // 0 .. 512*512-1
  int r = i & (HID - 1);
  int k = i >> 9;
  WhT[i] = Wh[(size_t)r * HID + k];                // WhT[k*512 + r], coalesced writes
}

// ---------------- K1: Xpre[t][n] = sum_k x[t][k] * Wx[n][k] ----------------
__global__ void rnn_pre_gemm(const float* __restrict__ x, const float* __restrict__ Wx,
                             float* __restrict__ Xpre) {
  const int lane = threadIdx.x & 31;
  const int wave = threadIdx.x >> 5;
  const int t0 = blockIdx.x * 16;
  const int n0 = (blockIdx.y * 4 + wave) * 16;
  const int rc = lane & 15;
  const int hi = lane >> 4;

  v8f c = {};
#pragma unroll
  for (int ko = 0; ko < IN_SZ; ko += 32) {
    v16bf a = frag_f32(x,  IN_SZ, t0 + rc, ko, hi);   // A: 16 timesteps x 32 k
    v16bf b = frag_f32(Wx, IN_SZ, n0 + rc, ko, hi);   // B: Wx row n is k-contig
    c = __builtin_amdgcn_wmma_f32_16x16x32_bf16(false, a, false, b,
                                                (short)0, c, false, false);
  }
  const int n  = n0 + (lane & 15);
  const int mb = t0 + 8 * (lane >> 4);
#pragma unroll
  for (int v = 0; v < 8; ++v)
    Xpre[(size_t)(mb + v) * HID + n] = c[v];
}

// ---------------- K2: sequential scan, one workgroup (one WGP) ----------------
__global__ void __launch_bounds__(1024)
rnn_scan(const float* __restrict__ WhT, const float* __restrict__ Xpre,
         const float* __restrict__ bh, __bf16* __restrict__ Hbf) {
  __shared__ float hs[HID];
  __shared__ float part[HID];
  const int tid = threadIdx.x;          // 0..1023
  const int r   = tid & (HID - 1);      // owned row
  const int kh  = tid >> 9;             // k-half: 0 or 1 (uniform per wave)

  if (tid < HID) hs[tid] = 0.0f;
  const float bias = bh[r];
  const float* __restrict__ wp = WhT + (size_t)(kh * 256) * HID + r; // coalesced
  const float* __restrict__ hp0 = hs + kh * 256;
  __syncthreads();

  for (int t = 0; t < SEQ_LEN; ++t) {
    __builtin_prefetch(Xpre + (size_t)(t + 1) * HID + r, 0, 0);
    float acc = 0.0f;
#pragma unroll 8
    for (int k = 0; k < 256; k += 4) {
      float4 hv = *(const float4*)(hp0 + k);          // LDS broadcast read
      acc += wp[(size_t)(k + 0) * HID] * hv.x;
      acc += wp[(size_t)(k + 1) * HID] * hv.y;
      acc += wp[(size_t)(k + 2) * HID] * hv.z;
      acc += wp[(size_t)(k + 3) * HID] * hv.w;
    }
    if (kh) part[r] = acc;
    __syncthreads();                                   // all reads of old h done
    if (tid < HID) {
      float hn = tanhf(acc + part[r] + Xpre[(size_t)t * HID + r] + bias);
      hs[r] = hn;
      Hbf[(size_t)t * HID + r] = f2bf(hn);
    }
    __syncthreads();                                   // new h visible
  }
}

// ---------------- K3: y[t][n] = sum_j H[t][j] * Wy[n][j] + by[n] ----------------
__global__ void rnn_post_gemm(const __bf16* __restrict__ Hbf, const float* __restrict__ Wy,
                              const float* __restrict__ by, float* __restrict__ y) {
  const int lane = threadIdx.x & 31;
  const int wave = threadIdx.x >> 5;
  const int t0 = blockIdx.x * 16;
  const int n0 = (blockIdx.y * 4 + wave) * 16;
  const int rc = lane & 15;
  const int hi = lane >> 4;

  v8f c = {};
#pragma unroll
  for (int ko = 0; ko < HID; ko += 32) {
    v16bf a = frag_bf16(Hbf, HID, t0 + rc, ko, hi);
    v16bf b = frag_f32 (Wy,  HID, n0 + rc, ko, hi);
    c = __builtin_amdgcn_wmma_f32_16x16x32_bf16(false, a, false, b,
                                                (short)0, c, false, false);
  }
  const int n    = n0 + (lane & 15);
  const int mb   = t0 + 8 * (lane >> 4);
  const float bb = by[n];
#pragma unroll
  for (int v = 0; v < 8; ++v)
    y[(size_t)(mb + v) * OUT_SZ + n] = c[v] + bb;
}

extern "C" void kernel_launch(void* const* d_in, const int* in_sizes, int n_in,
                              void* d_out, int out_size, void* d_ws, size_t ws_size,
                              hipStream_t stream) {
  (void)in_sizes; (void)n_in; (void)out_size; (void)ws_size;
  const float* x  = (const float*)d_in[0];
  const float* Wx = (const float*)d_in[1];
  const float* Wh = (const float*)d_in[2];
  const float* Wy = (const float*)d_in[3];
  const float* bh = (const float*)d_in[4];
  const float* by = (const float*)d_in[5];
  float* out = (float*)d_out;

  // Workspace layout (all 256B-aligned):
  //   [0,      1MB)   WhT   fp32 512x512
  //   [1MB,   65MB)   Xpre  fp32 32768x512
  //   [65MB,  97MB)   Hbf   bf16 32768x512
  char* ws = (char*)d_ws;
  float*  WhT  = (float*)ws;
  float*  Xpre = (float*)(ws + (size_t)(1u << 20));
  __bf16* Hbf  = (__bf16*)(ws + (size_t)(1u << 20) + (size_t)SEQ_LEN * HID * sizeof(float));

  wh_transpose<<<dim3((HID * HID) / 256), dim3(256), 0, stream>>>(Wh, WhT);
  rnn_pre_gemm<<<dim3(SEQ_LEN / 16, HID / 64), dim3(128), 0, stream>>>(x, Wx, Xpre);
  rnn_scan<<<dim3(1), dim3(1024), 0, stream>>>(WhT, Xpre, bh, Hbf);
  rnn_post_gemm<<<dim3(SEQ_LEN / 16, OUT_SZ / 64), dim3(128), 0, stream>>>(Hbf, Wy, by, out);
}
